// GINLayer_44315472560249
// MI455X (gfx1250) — matrix operand order repaired and checked by reading, not compiled, not run on hardware
//
#include <hip/hip_runtime.h>
#include <hip/hip_bf16.h>

typedef __attribute__((ext_vector_type(16))) __bf16 v16bf;
typedef __attribute__((ext_vector_type(8)))  float  v8f;

#define N_NODES 100000
#define N_EDGES 1600000
#define DIM     128
#define RPB     128          // rows per block in the MLP kernel
#define LSTRIDE 136          // LDS row stride (bf16 elems) to break bank conflicts
#define BN_EPS  1e-5f

// ---------------------------------------------------------------------------
// K0: zero neighbor-accumulator + column-stat accumulators in workspace
// ---------------------------------------------------------------------------
__global__ void zero_ws_kernel(float4* __restrict__ neigh4, int n4,
                               float* __restrict__ sums) {
    int i = blockIdx.x * blockDim.x + threadIdx.x;
    int stride = gridDim.x * blockDim.x;
    for (int j = i; j < n4; j += stride)
        neigh4[j] = make_float4(0.f, 0.f, 0.f, 0.f);
    if (i < 256) sums[i] = 0.f;
}

// ---------------------------------------------------------------------------
// K1: edge scatter  neigh[dst] += h[src]
// One wave (32 lanes) per edge; each lane moves 4 floats (float4 gather,
// 4x global_atomic_add_f32). h (51.2 MB) is L2-resident on MI455X (192 MB L2),
// so both the gather and the atomics run at L2 bandwidth, not HBM.
// ---------------------------------------------------------------------------
__global__ __launch_bounds__(256) void edge_scatter_kernel(
    const float* __restrict__ h, const int* __restrict__ src,
    const int* __restrict__ dst, float* __restrict__ neigh) {
    int gid = blockIdx.x * 256 + threadIdx.x;
    int e = gid >> 5;
    if (e >= N_EDGES) return;
    int lane = gid & 31;
    int s = src[e];      // uniform across the wave -> broadcast load
    int d = dst[e];
    float4 v = ((const float4*)h)[s * (DIM / 4) + lane];
    float* np = neigh + (size_t)d * DIM + lane * 4;
    atomicAdd(np + 0, v.x);
    atomicAdd(np + 1, v.y);
    atomicAdd(np + 2, v.z);
    atomicAdd(np + 3, v.w);
}

// ---------------------------------------------------------------------------
// K2: fused  x=(1+eps)h+neigh  ->  relu(x@W1+b1)  ->  @W2+b2  ->  *snorm
//     + per-column sum / sum-of-squares for batch norm.
// 256 threads = 8 waves; block owns 128 rows. W1,W2 transposed to bf16 in LDS.
// Each wave computes a 16x128 output strip with v_wmma_f32_16x16x32_bf16.
// ---------------------------------------------------------------------------
__global__ __launch_bounds__(256, 1) void gin_mlp_kernel(
    const float* __restrict__ h, const float* __restrict__ neigh,
    const float* __restrict__ snorm, const float* __restrict__ eps,
    const float* __restrict__ W1, const float* __restrict__ b1,
    const float* __restrict__ W2, const float* __restrict__ b2,
    float* __restrict__ xbuf, float* __restrict__ colsum,
    float* __restrict__ colsq) {
    __shared__ __bf16 Wt1[DIM * LSTRIDE];   // Wt1[n*LSTRIDE + k] = W1[k][n]
    __shared__ __bf16 Wt2[DIM * LSTRIDE];
    __shared__ __bf16 xs [RPB * LSTRIDE];   // activation tile (x, then y=relu(..))
    __shared__ float  snorms[RPB];
    __shared__ float  lsum[DIM];
    __shared__ float  lsq [DIM];

    const int tid  = threadIdx.x;
    const int row0 = blockIdx.x * RPB;
    const float ep1 = 1.0f + eps[0];

    // Stage weights (transposed, bf16) into LDS: 2 x 32 KB.
    for (int i = tid; i < DIM * DIM; i += 256) {
        int k = i >> 7, n = i & 127;
        Wt1[n * LSTRIDE + k] = (__bf16)W1[i];
        Wt2[n * LSTRIDE + k] = (__bf16)W2[i];
    }
    if (tid < DIM) { lsum[tid] = 0.f; lsq[tid] = 0.f; }
    if (tid < RPB) {
        int r = row0 + tid;
        snorms[tid] = (r < N_NODES) ? snorm[r] : 0.f;
    }
    // Stage x = (1+eps)*h + neigh as bf16 (zero-pad rows past N).
    for (int q = tid; q < RPB * (DIM / 4); q += 256) {
        int rl = q >> 5, c4 = q & 31;
        int r = row0 + rl;
        float4 x = make_float4(0.f, 0.f, 0.f, 0.f);
        if (r < N_NODES) {
            float4 h4 = ((const float4*)h)[r * (DIM / 4) + c4];
            float4 n4 = ((const float4*)neigh)[r * (DIM / 4) + c4];
            x.x = ep1 * h4.x + n4.x;
            x.y = ep1 * h4.y + n4.y;
            x.z = ep1 * h4.z + n4.z;
            x.w = ep1 * h4.w + n4.w;
        }
        __bf16* p = &xs[rl * LSTRIDE + c4 * 4];
        p[0] = (__bf16)x.x; p[1] = (__bf16)x.y;
        p[2] = (__bf16)x.z; p[3] = (__bf16)x.w;
    }
    __syncthreads();

    const int wave  = tid >> 5;
    const int lane  = tid & 31;
    const int half  = lane >> 4;   // lanes 16-31 hold the second K/M half
    const int ln    = lane & 15;
    const int rbase = wave * 16;   // this wave's 16-row strip

    // ---- GEMM1: y = relu(x @ W1 + b1), written back over xs (own rows only)
    // A fragment layout (16-bit A 16x32, ISA 7.12.2): lane L row M=L%16;
    // VGPR p<4 -> K=2p,2p+1 ; p>=4 -> K=16+2(p-4),.. ; +8 if lane>=16.
    v16bf afr[4];
#pragma unroll
    for (int kc = 0; kc < 4; ++kc) {
#pragma unroll
        for (int e = 0; e < 16; ++e) {
            const int p = e >> 1;
            const int k = kc * 32 + ((p < 4) ? (2 * p) : (16 + 2 * (p - 4)))
                          + 8 * half + (e & 1);
            afr[kc][e] = xs[(rbase + ln) * LSTRIDE + k];
        }
    }
#pragma unroll
    for (int nt = 0; nt < 8; ++nt) {
        v8f acc = {};
#pragma unroll
        for (int kc = 0; kc < 4; ++kc) {
            // B fragment (32x16 KxN): lane L col N=L%16, K = 8*(L/16)+e (+8 if e>=8).
            v16bf bfr;
#pragma unroll
            for (int e = 0; e < 16; ++e)
                bfr[e] = Wt1[(nt * 16 + ln) * LSTRIDE + kc * 32 + half * 8 + e + ((e >= 8) ? 8 : 0)];
            acc = __builtin_amdgcn_wmma_f32_16x16x32_bf16(
                false, afr[kc], false, bfr, (short)0, acc, false, false);
        }
        const int col  = nt * 16 + ln;
        const float bs = b1[col];
#pragma unroll
        for (int v = 0; v < 8; ++v) {
            float y = fmaxf(acc[v] + bs, 0.f);
            // D layout: VGPR v -> row v (lanes 0-15) / v+8 (lanes 16-31)
            xs[(rbase + v + 8 * half) * LSTRIDE + col] = (__bf16)y;
        }
    }
    __syncthreads();

    // ---- GEMM2: x2 = (y @ W2 + b2) * snorm ; store + column statistics
#pragma unroll
    for (int kc = 0; kc < 4; ++kc) {
#pragma unroll
        for (int e = 0; e < 16; ++e) {
            const int p = e >> 1;
            const int k = kc * 32 + ((p < 4) ? (2 * p) : (16 + 2 * (p - 4)))
                          + 8 * half + (e & 1);
            afr[kc][e] = xs[(rbase + ln) * LSTRIDE + k];
        }
    }
#pragma unroll
    for (int nt = 0; nt < 8; ++nt) {
        v8f acc = {};
#pragma unroll
        for (int kc = 0; kc < 4; ++kc) {
            v16bf bfr;
#pragma unroll
            for (int e = 0; e < 16; ++e)
                bfr[e] = Wt2[(nt * 16 + ln) * LSTRIDE + kc * 32 + half * 8 + e + ((e >= 8) ? 8 : 0)];
            acc = __builtin_amdgcn_wmma_f32_16x16x32_bf16(
                false, afr[kc], false, bfr, (short)0, acc, false, false);
        }
        const int col  = nt * 16 + ln;
        const float bs = b2[col];
        float s = 0.f, s2 = 0.f;
#pragma unroll
        for (int v = 0; v < 8; ++v) {
            const int rl = rbase + v + 8 * half;
            const int r  = row0 + rl;
            float val = (acc[v] + bs) * snorms[rl];
            if (r < N_NODES) {
                xbuf[(size_t)r * DIM + col] = val;
                s  += val;
                s2 += val * val;
            }
        }
        atomicAdd(&lsum[col], s);   // ds_add_f32
        atomicAdd(&lsq [col], s2);
    }
    __syncthreads();
    if (tid < DIM) {
        atomicAdd(&colsum[tid], lsum[tid]);
        atomicAdd(&colsq [tid], lsq [tid]);
    }
}

// ---------------------------------------------------------------------------
// K3: batch-norm (from global column stats) + relu + residual
// ---------------------------------------------------------------------------
__global__ __launch_bounds__(256) void bn_residual_kernel(
    const float* __restrict__ xbuf, const float* __restrict__ h,
    const float* __restrict__ colsum, const float* __restrict__ colsq,
    const float* __restrict__ gamma, const float* __restrict__ beta,
    float* __restrict__ out) {
    int q = blockIdx.x * 256 + threadIdx.x;
    if (q >= N_NODES * (DIM / 4)) return;
    int c0 = (q & 31) * 4;
    float4 x  = ((const float4*)xbuf)[q];
    float4 hh = ((const float4*)h)[q];
    float xv[4] = {x.x, x.y, x.z, x.w};
    float hv[4] = {hh.x, hh.y, hh.z, hh.w};
    float ov[4];
    const float invn = 1.0f / (float)N_NODES;
#pragma unroll
    for (int j = 0; j < 4; ++j) {
        int col = c0 + j;
        float mean = colsum[col] * invn;
        float var  = colsq[col] * invn - mean * mean;
        float rstd = rsqrtf(var + BN_EPS);
        float v = (xv[j] - mean) * rstd * gamma[col] + beta[col];
        ov[j] = hv[j] + fmaxf(v, 0.f);
    }
    ((float4*)out)[q] = make_float4(ov[0], ov[1], ov[2], ov[3]);
}

// ---------------------------------------------------------------------------
extern "C" void kernel_launch(void* const* d_in, const int* in_sizes, int n_in,
                              void* d_out, int out_size, void* d_ws, size_t ws_size,
                              hipStream_t stream) {
    const float* h     = (const float*)d_in[0];
    const float* snorm = (const float*)d_in[1];
    const float* eps   = (const float*)d_in[2];
    const float* W1    = (const float*)d_in[3];
    const float* b1    = (const float*)d_in[4];
    const float* W2    = (const float*)d_in[5];
    const float* b2    = (const float*)d_in[6];
    const float* gamma = (const float*)d_in[7];
    const float* beta  = (const float*)d_in[8];
    const int*   src   = (const int*)d_in[9];
    const int*   dst   = (const int*)d_in[10];
    float* out = (float*)d_out;

    char* ws = (char*)d_ws;
    const size_t nd_bytes = (size_t)N_NODES * DIM * sizeof(float);  // 51.2 MB
    float* neigh  = (float*)ws;
    float* xbuf   = (float*)(ws + nd_bytes);
    float* colsum = (float*)(ws + 2 * nd_bytes);
    float* colsq  = colsum + DIM;

    // K0: zero accumulators
    int n4 = N_NODES * (DIM / 4);
    zero_ws_kernel<<<(n4 + 255) / 256, 256, 0, stream>>>(
        (float4*)neigh, n4, colsum);

    // K1: edge scatter (one wave per edge)
    int scatter_threads = N_EDGES * 32;
    edge_scatter_kernel<<<scatter_threads / 256, 256, 0, stream>>>(
        h, src, dst, neigh);

    // K2: fused GIN update + 2-layer WMMA MLP + snorm + BN statistics
    int nblocks = (N_NODES + RPB - 1) / RPB;  // 782
    gin_mlp_kernel<<<nblocks, 256, 0, stream>>>(
        h, neigh, snorm, eps, W1, b1, W2, b2, xbuf, colsum, colsq);

    // K3: batch norm + relu + residual
    bn_residual_kernel<<<(n4 + 255) / 256, 256, 0, stream>>>(
        xbuf, h, colsum, colsq, gamma, beta, out);
}
